// ECMNet_17497696763958
// MI455X (gfx1250) — matrix operand order repaired and checked by reading, not compiled, not run on hardware
//
#include <hip/hip_runtime.h>
#include <hip/hip_bf16.h>
#include <cstdint>
#include <cstddef>

// Problem constants (match reference)
#define BB 2
#define FF 2048
#define TT 750
#define CC 20
#define GG 3
#define KTOP 93
#define KKD 6144  // 3 * 2048 im2col depth

typedef __attribute__((ext_vector_type(16))) __bf16        v16bf;
typedef __attribute__((ext_vector_type(8)))  float         v8f;
typedef __attribute__((ext_vector_type(4)))  unsigned int  v4u;
typedef __attribute__((ext_vector_type(4)))  int           v4i;

union FragU { v16bf f; v4u u[2]; };

#if __has_builtin(__builtin_amdgcn_global_load_async_to_lds_b128) && \
    __has_builtin(__builtin_amdgcn_s_wait_asynccnt)
#define USE_ASYNC 1
typedef __attribute__((address_space(1))) v4i* as1_v4i;
typedef __attribute__((address_space(3))) v4i* as3_v4i;
#else
#define USE_ASYNC 0
#endif

__device__ __forceinline__ void cp16(const unsigned short* g, unsigned short* l) {
#if USE_ASYNC
  __builtin_amdgcn_global_load_async_to_lds_b128(
      (as1_v4i)(unsigned long long)(uintptr_t)g,
      (as3_v4i)(unsigned int)(uintptr_t)l, 0, 0);
#else
  *(v4u*)l = *(const v4u*)g;
#endif
}

__device__ __forceinline__ unsigned short f2bf(float x) {
  unsigned int u = __builtin_bit_cast(unsigned int, x);
  u += 0x7fffu + ((u >> 16) & 1u);
  return (unsigned short)(u >> 16);
}
__device__ __forceinline__ float bf2f(unsigned short h) {
  unsigned int u = ((unsigned int)h) << 16;
  return __builtin_bit_cast(float, u);
}

// ---------------------------------------------------------------------------
__global__ __launch_bounds__(256) void zero_k(unsigned int* __restrict__ p, int n) {
  int id = blockIdx.x * 256 + threadIdx.x;
  if (id < n) p[id] = 0u;
}

// Pack x (B,F,T) fp32 -> xT [b*T + t][ci] bf16   (ci contiguous)
__global__ __launch_bounds__(256) void pack_x_k(const float* __restrict__ x,
                                                unsigned short* __restrict__ xT) {
  int id = blockIdx.x * 256 + threadIdx.x;  // over BB*TT*FF
  int f = id & (FF - 1);
  int colid = id >> 11;  // b*TT + t
  int t = colid % TT;
  int b = colid / TT;
  xT[id] = f2bf(x[((size_t)(b * FF + f)) * TT + t]);
}

// Pack w (Cout,Cin,3) fp32 -> Wpk [co][k*2048+ci] bf16
__global__ __launch_bounds__(256) void pack_w_k(const float* __restrict__ w,
                                                unsigned short* __restrict__ Wp) {
  int id = blockIdx.x * 256 + threadIdx.x;  // over 2048*6144
  int ci = id & (FF - 1);
  int r = id >> 11;
  int k = r % 3;
  int co = r / 3;
  Wp[id] = f2bf(w[((size_t)(co * FF + ci)) * 3 + k]);
}

// ---------------------------------------------------------------------------
// k=3 pad=1 conv as WMMA bf16 GEMM, async-LDS staged + double-buffered.
//   X  : [nimg*L][2048] bf16 (GEMM B, ci contiguous)
//   W  : [2048][6144] bf16   (GEMM A, im2col rows)
//   Zr : 2048 bf16 zeros (branchless padding target)
//   Out: [nimg*L][2048] bf16 (bias + leaky-relu fused)
// Grid: (2048/128, ceil(NC/128)), 256 threads.
// 8 waves = 4(M) x 2(N); wave tile 32x64 = 2x4 WMMA fragments.
// ---------------------------------------------------------------------------
#define AST 40  // LDS row stride (ushorts): 32 data + 8 pad (spreads banks)
#define BST 40

__global__ __launch_bounds__(256)
void conv3_wmma(const unsigned short* __restrict__ X,
                const unsigned short* __restrict__ W,
                const float* __restrict__ bias,
                const unsigned short* __restrict__ Zr,
                unsigned short* __restrict__ Out,
                int L, int NC) {
  __shared__ unsigned short lA[2][128 * AST];  // 2 x 10240 B
  __shared__ unsigned short lB[2][128 * BST];  // 2 x 10240 B

  const int tid  = threadIdx.x;
  const int lane = tid & 31;
  const int wv   = tid >> 5;  // 0..7
  const int mw   = wv >> 1;   // 0..3  (M wave coord, 32 rows each)
  const int nw   = wv & 1;    // 0..1  (N wave coord, 64 cols each)
  const int hi   = lane >> 4;
  const int ln16 = lane & 15;

  const int co0  = blockIdx.x * 128;
  const int col0 = blockIdx.y * 128;

  // ---- staging roles (branchless; invalid taps read the zero row) ----
  // A: 512 16B-chunks (128 rows x 32 K); B: 512 16B-chunks (128 cols x 32 K).
  const unsigned short* aPtr[2];
  int aL[2];
  const unsigned short* bPtr[2][3];
  int bL[2];
#pragma unroll
  for (int part = 0; part < 2; ++part) {
    int chunk = tid + 256 * part;
    int row = chunk >> 2, q = chunk & 3;
    aPtr[part] = W + (size_t)(co0 + row) * KKD + q * 8;
    aL[part] = row * AST + q * 8;

    int cg = col0 + row;  // same decomposition for B: col = chunk>>2
    bool inc = cg < NC;
    int lp = (inc ? cg : 0) % L;
#pragma unroll
    for (int k = 0; k < 3; ++k) {
      int le = lp + k - 1;
      bool v = inc && (le >= 0) && (le < L);
      bPtr[part][k] = v ? (X + (size_t)(cg + k - 1) * FF + q * 8) : (Zr + q * 8);
    }
    bL[part] = row * BST + q * 8;
  }

  // ---- fragment read offsets (per documented 16-bit A/B VGPR layouts) ----
  const int aRd = (mw * 32 + ln16) * AST + hi * 8;   // +mi*16*AST ; u[1] at +16
  const int bRd = (nw * 64 + ln16) * BST + hi * 8;   // +ni*16*BST ; u[1] at +16

  v8f acc[2][4];
#pragma unroll
  for (int mi = 0; mi < 2; ++mi)
#pragma unroll
    for (int ni = 0; ni < 4; ++ni) acc[mi][ni] = {};

  auto stage = [&](int buf, int kk0) {
    const int k = kk0 >> 11;         // conv tap (2048 % 32 == 0)
    const int c0 = kk0 & (FF - 1);
    cp16(aPtr[0] + kk0, &lA[buf][aL[0]]);
    cp16(aPtr[1] + kk0, &lA[buf][aL[1]]);
    cp16(bPtr[0][k] + c0, &lB[buf][bL[0]]);
    cp16(bPtr[1][k] + c0, &lB[buf][bL[1]]);
  };

  // ---- stage first tile ----
  stage(0, 0);

  int buf = 0;
  for (int kk0 = 0; kk0 < KKD; kk0 += 32) {
    // stage next tile into other buffer
    if (kk0 + 32 < KKD) {
      stage(buf ^ 1, kk0 + 32);
#if USE_ASYNC
      __builtin_amdgcn_s_wait_asynccnt(4);  // current tile done; next 4 in flight
#endif
    } else {
#if USE_ASYNC
      __builtin_amdgcn_s_wait_asynccnt(0);
#endif
    }
    __syncthreads();  // current tile visible to all waves

    FragU A[2], Bf[4];
#pragma unroll
    for (int mi = 0; mi < 2; ++mi) {
      const unsigned short* p = &lA[buf][aRd + mi * 16 * AST];
      A[mi].u[0] = *(const v4u*)(p);       // K hi*8 .. +7      -> VGPR 0-3
      A[mi].u[1] = *(const v4u*)(p + 16);  // K 16+hi*8 .. +7   -> VGPR 4-7
    }
#pragma unroll
    for (int ni = 0; ni < 4; ++ni) {
      const unsigned short* p = &lB[buf][bRd + ni * 16 * BST];
      Bf[ni].u[0] = *(const v4u*)(p);       // K hi*8 .. +7      -> VGPR 0-3
      Bf[ni].u[1] = *(const v4u*)(p + 16);  // K 16+hi*8 .. +7   -> VGPR 4-7
    }

#pragma unroll
    for (int mi = 0; mi < 2; ++mi)
#pragma unroll
      for (int ni = 0; ni < 4; ++ni)
        acc[mi][ni] = __builtin_amdgcn_wmma_f32_16x16x32_bf16(
            false, A[mi].f, false, Bf[ni].f, (short)0, acc[mi][ni], false, false);

    __syncthreads();  // all reads of current buffer done before it is re-staged
    buf ^= 1;
  }

  // ---- epilogue: D layout -> lane ln16 = column, VGPR v = row v + hi*8 ----
#pragma unroll
  for (int mi = 0; mi < 2; ++mi) {
#pragma unroll
    for (int ni = 0; ni < 4; ++ni) {
      const int col = col0 + nw * 64 + ni * 16 + ln16;
      if (col >= NC) continue;
      const int cob = co0 + mw * 32 + mi * 16 + hi * 8;
      unsigned short hv[8];
#pragma unroll
      for (int v = 0; v < 8; ++v) {
        float f = acc[mi][ni][v] + bias[cob + v];
        f = (f > 0.f) ? f : 0.01f * f;  // LeakyReLU(0.01)
        hv[v] = f2bf(f);
      }
      v4u pk;
      pk.x = (unsigned)hv[0] | ((unsigned)hv[1] << 16);
      pk.y = (unsigned)hv[2] | ((unsigned)hv[3] << 16);
      pk.z = (unsigned)hv[4] | ((unsigned)hv[5] << 16);
      pk.w = (unsigned)hv[6] | ((unsigned)hv[7] << 16);
      *(v4u*)(Out + (size_t)col * FF + cob) = pk;
    }
  }
}

// ---------------------------------------------------------------------------
// seq_cas[b][c][t] = dot(wc[c], h2[col])   (1x1 conv, no bias)
// ---------------------------------------------------------------------------
__global__ __launch_bounds__(256)
void cls_seq_k(const unsigned short* __restrict__ H, const float* __restrict__ wc,
               float* __restrict__ seq) {
  const int col = blockIdx.x;  // b*TT + t
  const int tid = threadIdx.x;
  float acc[CC];
#pragma unroll
  for (int c = 0; c < CC; ++c) acc[c] = 0.f;
  for (int ci = tid; ci < FF; ci += 256) {
    float xv = bf2f(H[(size_t)col * FF + ci]);
#pragma unroll
    for (int c = 0; c < CC; ++c) acc[c] += xv * wc[c * FF + ci];
  }
  __shared__ float red[256];
  const int b = col / TT, t = col % TT;
  for (int c = 0; c < CC; ++c) {
    red[tid] = acc[c];
    __syncthreads();
    for (int s = 128; s > 0; s >>= 1) {
      if (tid < s) red[tid] += red[tid + s];
      __syncthreads();
    }
    if (tid == 0) seq[((size_t)(b * CC + c)) * TT + t] = red[0];
    __syncthreads();
  }
}

// ---------------------------------------------------------------------------
// score_pre row = mean(top-93 of seq row). Bitonic sort 1024 (pad -inf).
// ---------------------------------------------------------------------------
__global__ __launch_bounds__(256)
void topk_mean_k(const float* __restrict__ seq, float* __restrict__ outp) {
  const int row = blockIdx.x;  // b*CC + c
  const int tid = threadIdx.x;
  __shared__ float s[1024];
  for (int i = tid; i < 1024; i += 256)
    s[i] = (i < TT) ? seq[(size_t)row * TT + i] : -3.0e38f;
  __syncthreads();
  for (int k = 2; k <= 1024; k <<= 1) {
    for (int j = k >> 1; j > 0; j >>= 1) {
      for (int base = 0; base < 1024; base += 256) {
        int i = base + tid;
        int ixj = i ^ j;
        if (ixj > i) {
          bool up = ((i & k) == 0);
          float a = s[i], b2 = s[ixj];
          if ((a > b2) == up) { s[i] = b2; s[ixj] = a; }  // ascending
        }
      }
      __syncthreads();
    }
  }
  __shared__ float red[256];
  float v = 0.f;
  for (int i = tid; i < KTOP; i += 256) v += s[1024 - KTOP + i];
  red[tid] = v;
  __syncthreads();
  for (int st = 128; st > 0; st >>= 1) {
    if (tid < st) red[tid] += red[tid + st];
    __syncthreads();
  }
  if (tid == 0) outp[row] = red[0] / (float)KTOP;
}

// ---------------------------------------------------------------------------
// wa = sigmoid(watt @ seq + batt)  (1x1 over C channels)
// ---------------------------------------------------------------------------
__global__ __launch_bounds__(256)
void att_sig_k(const float* __restrict__ seq, const float* __restrict__ watt,
               const float* __restrict__ batt, float* __restrict__ wa) {
  int idx = blockIdx.x * 256 + threadIdx.x;
  if (idx >= BB * CC * TT) return;
  int t = idx % TT;
  int r = idx / TT;
  int c = r % CC;
  int b = r / CC;
  float s = batt[c];
#pragma unroll
  for (int c2 = 0; c2 < CC; ++c2)
    s += watt[c * CC + c2] * seq[((size_t)(b * CC + c2)) * TT + t];
  wa[idx] = 1.f / (1.f + __expf(-s));
}

// ---------------------------------------------------------------------------
// feat_agg[bc][g][f] = sum_j wa[b,c,3j+g] * x[b,f,3j+g]   -> bf16 [img*3+g][f]
// ---------------------------------------------------------------------------
__global__ __launch_bounds__(256)
void feat_agg_k(const unsigned short* __restrict__ xT, const float* __restrict__ wa,
                unsigned short* __restrict__ outF) {
  const int bx = blockIdx.x;  // bc*3+g
  const int g = bx % 3;
  const int bc = bx / 3;
  const int b = bc / CC;
  const int tid = threadIdx.x;
  __shared__ float wsh[250];
  for (int j = tid; j < 250; j += 256) wsh[j] = wa[(size_t)bc * TT + 3 * j + g];
  __syncthreads();
  const int f = blockIdx.y * 256 + tid;
  float acc = 0.f;
  for (int j = 0; j < 250; ++j)
    acc += wsh[j] * bf2f(xT[((size_t)(b * TT + 3 * j + g)) * FF + f]);
  outF[(size_t)bx * FF + f] = f2bf(acc);
}

// ---------------------------------------------------------------------------
// background agg[((b*C+c)*C+a)][g][f] = sum_j wa[b,a,t]*(1-wa[b,c,t])*x[b,f,t]
// ---------------------------------------------------------------------------
__global__ __launch_bounds__(256)
void bg_agg_k(const unsigned short* __restrict__ xT, const float* __restrict__ wa,
              unsigned short* __restrict__ outF) {
  const int bx = blockIdx.x;  // (((b*C+c)*C)+a)*3+g
  const int g = bx % 3;
  int r = bx / 3;
  const int a = r % CC;
  r /= CC;
  const int c = r % CC;
  const int b = r / CC;
  const int tid = threadIdx.x;
  __shared__ float wsh[250];
  for (int j = tid; j < 250; j += 256) {
    int t = 3 * j + g;
    float waA = wa[((size_t)(b * CC + a)) * TT + t];
    float waC = wa[((size_t)(b * CC + c)) * TT + t];
    wsh[j] = waA * (1.f - waC);
  }
  __syncthreads();
  const int f = blockIdx.y * 256 + tid;
  float acc = 0.f;
  for (int j = 0; j < 250; ++j)
    acc += wsh[j] * bf2f(xT[((size_t)(b * TT + 3 * j + g)) * FF + f]);
  outF[(size_t)bx * FF + f] = f2bf(acc);
}

// ---------------------------------------------------------------------------
// grouped classifier + mean over g: score[img] = mean_l dot(wc[img%C], H[img*3+l])
// ---------------------------------------------------------------------------
__global__ __launch_bounds__(256)
void grouped_cls_k(const unsigned short* __restrict__ H, const float* __restrict__ wc,
                   float* __restrict__ sc) {
  const int img = blockIdx.x;
  const int cls = img % CC;
  const int tid = threadIdx.x;
  float a0 = 0.f, a1 = 0.f, a2 = 0.f;
  for (int ci = tid; ci < FF; ci += 256) {
    float w = wc[cls * FF + ci];
    a0 += w * bf2f(H[((size_t)img * 3 + 0) * FF + ci]);
    a1 += w * bf2f(H[((size_t)img * 3 + 1) * FF + ci]);
    a2 += w * bf2f(H[((size_t)img * 3 + 2) * FF + ci]);
  }
  __shared__ float red[256];
  red[tid] = a0 + a1 + a2;
  __syncthreads();
  for (int s = 128; s > 0; s >>= 1) {
    if (tid < s) red[tid] += red[tid + s];
    __syncthreads();
  }
  if (tid == 0) sc[img] = red[0] / 3.0f;
}

// ---------------------------------------------------------------------------
// collect/diag/softmax combine. out layout: [pre(40) | post(40) | agg(80)]
// ---------------------------------------------------------------------------
__global__ __launch_bounds__(64)
void combine_k(const float* __restrict__ sp, const float* __restrict__ sc,
               const int* __restrict__ lab, float* __restrict__ out) {
  const int tid = threadIdx.x;
  __shared__ float cat[2 * CC];
  __shared__ float mx, ssum;
  for (int b = 0; b < BB; ++b) {
    if (tid < CC) {
      out[BB * CC + b * CC + tid] = sp[b * CC + tid];  // score_post passthrough
      const int a = tid;
      float ngt = 0.f, acc = 0.f;
      for (int c = 0; c < CC; ++c) {
        float m = (float)lab[b * CC + c];
        ngt += m;
        acc += m * sc[((size_t)(b * CC + c)) * CC + a];
      }
      if (ngt < 1.f) ngt = 1.f;
      float col = acc / ngt;
      if (lab[b * CC + a] > 0) col = sc[((size_t)(b * CC + a)) * CC + a];  // diag
      cat[a] = sp[b * CC + a];
      cat[CC + a] = col;
    }
    __syncthreads();
    if (tid == 0) {
      float m = -3.0e38f;
      for (int i = 0; i < 2 * CC; ++i) m = m > cat[i] ? m : cat[i];
      mx = m;
    }
    __syncthreads();
    if (tid < 2 * CC) cat[tid] = expf(cat[tid] - mx);
    __syncthreads();
    if (tid == 0) {
      float s = 0.f;
      for (int i = 0; i < 2 * CC; ++i) s += cat[i];
      ssum = s;
    }
    __syncthreads();
    if (tid < 2 * CC) out[2 * BB * CC + b * 2 * CC + tid] = cat[tid] / ssum;
    __syncthreads();
  }
}

// ---------------------------------------------------------------------------
extern "C" void kernel_launch(void* const* d_in, const int* in_sizes, int n_in,
                              void* d_out, int out_size, void* d_ws, size_t ws_size,
                              hipStream_t stream) {
  const float* x    = (const float*)d_in[0];
  const float* w1   = (const float*)d_in[1];
  const float* b1   = (const float*)d_in[2];
  const float* w2   = (const float*)d_in[3];
  const float* b2   = (const float*)d_in[4];
  const float* wc   = (const float*)d_in[5];
  const float* watt = (const float*)d_in[6];
  const float* batt = (const float*)d_in[7];
  const int*   lab  = (const int*)d_in[8];
  float* out = (float*)d_out;

  char* wsbase = (char*)d_ws;
  size_t off = 0;
  auto alloc = [&](size_t bytes) -> char* {
    char* p = wsbase + off;
    off = (off + bytes + 255) & ~(size_t)255;
    return p;
  };

  unsigned short* xT   = (unsigned short*)alloc((size_t)BB * TT * FF * 2);
  unsigned short* W1p  = (unsigned short*)alloc((size_t)FF * KKD * 2);
  unsigned short* W2p  = (unsigned short*)alloc((size_t)FF * KKD * 2);
  unsigned short* h1   = (unsigned short*)alloc((size_t)BB * TT * FF * 2);
  unsigned short* h2   = (unsigned short*)alloc((size_t)BB * TT * FF * 2);
  float*          seq  = (float*)alloc((size_t)BB * CC * TT * 4);
  float*          wa   = (float*)alloc((size_t)BB * CC * TT * 4);
  unsigned short* fA   = (unsigned short*)alloc((size_t)BB * CC * 3 * FF * 2);
  unsigned short* hp1  = (unsigned short*)alloc((size_t)BB * CC * 3 * FF * 2);
  unsigned short* aggB = (unsigned short*)alloc((size_t)BB * CC * CC * 3 * FF * 2);
  unsigned short* hb1  = (unsigned short*)alloc((size_t)BB * CC * CC * 3 * FF * 2);
  float*          spst = (float*)alloc((size_t)BB * CC * 4);
  float*          scbg = (float*)alloc((size_t)BB * CC * CC * 4);
  unsigned short* Zr   = (unsigned short*)alloc((size_t)FF * 2);  // zero row
  unsigned short* hp2  = fA;    // alias: fA consumed before hp2 written
  unsigned short* hb2  = aggB;  // alias: aggB consumed before hb2 written

  // 0. zero pad row
  zero_k<<<(FF / 2 + 255) / 256, 256, 0, stream>>>((unsigned int*)Zr, FF / 2);

  // 1. pack inputs to bf16
  pack_x_k<<<(BB * TT * FF) / 256, 256, 0, stream>>>(x, xT);
  pack_w_k<<<((size_t)FF * KKD) / 256, 256, 0, stream>>>(w1, W1p);
  pack_w_k<<<((size_t)FF * KKD) / 256, 256, 0, stream>>>(w2, W2p);

  // 2. main stream: base (two convs), then classifier
  {
    int NC = BB * TT;  // 1500
    dim3 g(FF / 128, (NC + 127) / 128);
    conv3_wmma<<<g, 256, 0, stream>>>(xT, W1p, b1, Zr, h1, TT, NC);
    conv3_wmma<<<g, 256, 0, stream>>>(h1, W2p, b2, Zr, h2, TT, NC);
  }
  cls_seq_k<<<BB * TT, 256, 0, stream>>>(h2, wc, seq);

  // 3. score_pre -> out[0..39]
  topk_mean_k<<<BB * CC, 256, 0, stream>>>(seq, out);

  // 4. attention weights
  att_sig_k<<<(BB * CC * TT + 255) / 256, 256, 0, stream>>>(seq, watt, batt, wa);

  // 5. foreground aggregation + post stream
  feat_agg_k<<<dim3(BB * CC * 3, FF / 256), 256, 0, stream>>>(xT, wa, fA);
  {
    int NC = BB * CC * 3;  // 120
    dim3 g(FF / 128, (NC + 127) / 128);
    conv3_wmma<<<g, 256, 0, stream>>>(fA, W1p, b1, Zr, hp1, 3, NC);
    conv3_wmma<<<g, 256, 0, stream>>>(hp1, W2p, b2, Zr, hp2, 3, NC);
  }
  grouped_cls_k<<<BB * CC, 256, 0, stream>>>(hp2, wc, spst);

  // 6. background stream (training branch)
  bg_agg_k<<<dim3(BB * CC * CC * 3, FF / 256), 256, 0, stream>>>(xT, wa, aggB);
  {
    int NC = BB * CC * CC * 3;  // 2400
    dim3 g(FF / 128, (NC + 127) / 128);
    conv3_wmma<<<g, 256, 0, stream>>>(aggB, W1p, b1, Zr, hb1, 3, NC);
    conv3_wmma<<<g, 256, 0, stream>>>(hb1, W2p, b2, Zr, hb2, 3, NC);
  }
  grouped_cls_k<<<BB * CC * CC, 256, 0, stream>>>(hb2, wc, scbg);

  // 7. combine: score_post -> out[40..79], score_agg -> out[80..159]
  combine_k<<<1, 64, 0, stream>>>(spst, scbg, lab, out);
}
